// TransformerLM_9655086482191
// MI455X (gfx1250) — matrix operand order repaired and checked by reading, not compiled, not run on hardware
//
#include <hip/hip_runtime.h>
#include <hip/hip_bf16.h>
#include <math.h>
#include <stdint.h>

// ---------------------------------------------------------------------------
// TransformerLM forward for MI455X (gfx1250, wave32, WMMA + async-LDS copies).
// Dims: V=2000 B=1024 T=64 D=128 H=4 HD=32 L=2 FF=512, BT=65536.
// GEMM chain is f16-native: activations stored f16, weights pre-transposed to
// f16 [N,K]; tiles move global->LDS via global_load_async_to_lds_b128.
// ---------------------------------------------------------------------------

typedef __attribute__((ext_vector_type(16))) _Float16 v16h;
typedef __attribute__((ext_vector_type(8)))  _Float16 v8h;
typedef __attribute__((ext_vector_type(4)))  _Float16 v4h;
typedef __attribute__((ext_vector_type(8)))  float    v8f;

#define BT_   65536
#define D_    128
#define H_    4
#define HD_   32
#define T_    64
#define FF_   512
#define V_    2000
#define L_    2

// ---- helpers --------------------------------------------------------------

__device__ __forceinline__ v16h cat8(v8h lo, v8h hi) {
    return __builtin_shufflevector(lo, hi, 0, 1, 2, 3, 4, 5, 6, 7,
                                           8, 9, 10, 11, 12, 13, 14, 15);
}

// Async 16-byte copy global -> LDS (CDNA5, tracked by ASYNCcnt).
// lds_byte = low 32 bits of the flat shared-pointer (= LDS offset).
__device__ __forceinline__ void async_b128(uint32_t lds_byte, const void* g) {
    asm volatile("global_load_async_to_lds_b128 %0, %1, off"
                 :: "v"(lds_byte), "v"(g)
                 : "memory");
}
__device__ __forceinline__ void wait_async0() {
    asm volatile("s_wait_asynccnt 0x0" ::: "memory");
}
__device__ __forceinline__ uint32_t lds_addr(const void* p) {
    return (uint32_t)(uintptr_t)p;   // shared aperture: low 32 bits = LDS offset
}

// ---- WMMA fragment loaders (layouts per CDNA5 ISA 7.12.2, wave32) ---------
// Each loader is two contiguous 16-byte LDS reads -> ds_load_b128 pairs.

// A fragment: 16x32 f16, lane holds row M=(lane&15).
// lanes 0-15: halves 0..7 -> K=0..7, 8..15 -> K=16..23; lanes 16-31: K+8.
__device__ __forceinline__ v16h load_frag_a(const _Float16* lds, int row0,
                                            int ldm, int lane) {
    const _Float16* p = lds + (row0 + (lane & 15)) * ldm + ((lane >> 4) << 3);
    v8h lo = *(const v8h*)(p);
    v8h hi = *(const v8h*)(p + 16);
    return cat8(lo, hi);
}

// B fragment from a K-contiguous (transposed) tile: src[n][k].
// lanes 0-15 hold col N=(lane&15), K=0..7,16..23; lanes 16-31: K+8.
__device__ __forceinline__ v16h load_frag_bT(const _Float16* lds, int col0,
                                             int ld, int lane) {
    const _Float16* p = lds + (col0 + (lane & 15)) * ld + ((lane >> 4) << 3);
    v8h lo = *(const v8h*)(p);
    v8h hi = *(const v8h*)(p + 16);
    return cat8(lo, hi);
}

// ---------------------------------------------------------------------------
// Weight prep: f32 [K,N] row-major  ->  f16 [N,K] (transposed).
// Tiny (<=256K elems per matrix); runs once per launch.
// ---------------------------------------------------------------------------
__global__ __launch_bounds__(256) void transpose_cvt_kernel(const float* __restrict__ W,
                                                            _Float16* __restrict__ Wt,
                                                            int K, int N) {
    int idx = blockIdx.x * 256 + threadIdx.x;
    if (idx < K * N) {
        int k = idx / N, n = idx - k * N;
        Wt[(size_t)n * K + k] = (_Float16)W[idx];
    }
}

// ---------------------------------------------------------------------------
// Embedding: x[row, :] = tok_emb[idx[row], :] + pos_emb[row % T, :]
// ---------------------------------------------------------------------------
__global__ __launch_bounds__(256) void embed_kernel(const int* __restrict__ idx,
                                                    const float* __restrict__ tok,
                                                    const float* __restrict__ pos,
                                                    float* __restrict__ x) {
    size_t row = (size_t)blockIdx.x * 2 + (threadIdx.x >> 7);
    int d = threadIdx.x & 127;
    int t = (int)(row & (T_ - 1));
    int id = idx[row];
    x[row * D_ + d] = tok[(size_t)id * D_ + d] + pos[(size_t)t * D_ + d];
}

// ---------------------------------------------------------------------------
// LayerNorm (f32 in -> f16 out): one wave per row, shfl reduction.
// ---------------------------------------------------------------------------
__global__ __launch_bounds__(256) void layernorm_kernel(const float* __restrict__ x,
                                                        const float* __restrict__ g,
                                                        const float* __restrict__ bta,
                                                        _Float16* __restrict__ out) {
    int lane = threadIdx.x & 31;
    int w = threadIdx.x >> 5;
    size_t row = (size_t)blockIdx.x * 8 + w;
    const float4 v = ((const float4*)(x + row * D_))[lane];
    float s  = v.x + v.y + v.z + v.w;
    float s2 = v.x * v.x + v.y * v.y + v.z * v.z + v.w * v.w;
#pragma unroll
    for (int off = 16; off > 0; off >>= 1) {
        s  += __shfl_xor(s,  off, 32);
        s2 += __shfl_xor(s2, off, 32);
    }
    float mean = s * (1.0f / 128.0f);
    float var  = s2 * (1.0f / 128.0f) - mean * mean;
    float rs   = rsqrtf(var + 1e-5f);
    float4 gv = ((const float4*)g)[lane];
    float4 bv = ((const float4*)bta)[lane];
    v4h o;
    o[0] = (_Float16)((v.x - mean) * rs * gv.x + bv.x);
    o[1] = (_Float16)((v.y - mean) * rs * gv.y + bv.y);
    o[2] = (_Float16)((v.z - mean) * rs * gv.z + bv.z);
    o[3] = (_Float16)((v.w - mean) * rs * gv.w + bv.w);
    ((v4h*)(out + row * D_))[lane] = o;
}

// ---------------------------------------------------------------------------
// WMMA GEMM:  out = [relu](A @ W + bias) [+ residual]
// A: f16 [M,K]; Wt: f16 [N,K] (pre-transposed). 64x64 C tile per 8-wave block.
// Tiles staged via global_load_async_to_lds_b128 (one b128 per thread per
// tile), double-buffered with a single barrier per K-step.
// Output either f16 (outh) or f32 (+bias/relu/residual) (outf).
// ---------------------------------------------------------------------------
__global__ __launch_bounds__(256) void gemm_wmma_kernel(const _Float16* __restrict__ A,
                                                        const _Float16* __restrict__ Wt,
                                                        const float* __restrict__ bias,
                                                        const float* __restrict__ residual,
                                                        float* __restrict__ outf,
                                                        _Float16* __restrict__ outh,
                                                        int M, int N, int K, int relu) {
    __shared__ __align__(16) _Float16 sA[2][64 * 32];   // [row][k]
    __shared__ __align__(16) _Float16 sBt[2][64 * 40];  // [col][k], padded stride

    int tid = threadIdx.x;
    int lane = tid & 31;
    int w = tid >> 5;
    int m0 = blockIdx.x * 64;
    int n0 = blockIdx.y * 64;
    int wm = w >> 1;           // 0..3  (row strip)
    int wn = w & 1;            // 0..1  (col half)

    // staging: one 16B chunk per thread per tile
    int ar = tid >> 2, ac = (tid & 3) * 8;    // A: row ar, k-halves ac..ac+7
    int bn = tid >> 2, bk = (tid & 3) * 8;    // B: col bn, k-halves bk..bk+7
    int bcol = n0 + bn; if (bcol >= N) bcol = N - 1;   // clamp (stores guarded)

    uint32_t ldsA[2], ldsB[2];
    ldsA[0] = lds_addr(&sA[0][ar * 32 + ac]);
    ldsA[1] = lds_addr(&sA[1][ar * 32 + ac]);
    ldsB[0] = lds_addr(&sBt[0][bn * 40 + bk]);
    ldsB[1] = lds_addr(&sBt[1][bn * 40 + bk]);
    const _Float16* gA = A + (size_t)(m0 + ar) * K + ac;
    const _Float16* gB = Wt + (size_t)bcol * K + bk;

    v8f acc0 = {};
    v8f acc1 = {};

    // prologue: tile 0 -> buffer 0
    async_b128(ldsA[0], gA);
    async_b128(ldsB[0], gB);

    for (int k0 = 0; k0 < K; k0 += 32) {
        int buf = (k0 >> 5) & 1;
        wait_async0();
        __syncthreads();
        if (k0 + 32 < K) {           // issue next tile AFTER barrier (race-free)
            async_b128(ldsA[buf ^ 1], gA + k0 + 32);
            async_b128(ldsB[buf ^ 1], gB + k0 + 32);
        }
        v16h a  = load_frag_a(sA[buf], wm * 16, 32, lane);
        v16h b0 = load_frag_bT(sBt[buf], wn * 32, 40, lane);
        v16h b1 = load_frag_bT(sBt[buf], wn * 32 + 16, 40, lane);
        acc0 = __builtin_amdgcn_wmma_f32_16x16x32_f16(false, a, false, b0,
                                                      (short)0, acc0, false, false);
        acc1 = __builtin_amdgcn_wmma_f32_16x16x32_f16(false, a, false, b1,
                                                      (short)0, acc1, false, false);
    }

    // epilogue: C/D layout -> lane col = lane&15, rows j + 8*(lane>=16)
    int colA = n0 + wn * 32 + (lane & 15);
    int colB = colA + 16;
    int rbase = m0 + wm * 16 + ((lane >> 4) << 3);
#pragma unroll
    for (int j = 0; j < 8; ++j) {
        int row = rbase + j;
        if (colA < N) {
            float v0 = acc0[j];
            if (bias) v0 += bias[colA];
            if (relu) v0 = fmaxf(v0, 0.0f);
            if (outh) {
                outh[(size_t)row * N + colA] = (_Float16)v0;
            } else {
                if (residual) v0 += residual[(size_t)row * N + colA];
                outf[(size_t)row * N + colA] = v0;
            }
        }
        if (colB < N) {
            float v1 = acc1[j];
            if (bias) v1 += bias[colB];
            if (relu) v1 = fmaxf(v1, 0.0f);
            if (outh) {
                outh[(size_t)row * N + colB] = (_Float16)v1;
            } else {
                if (residual) v1 += residual[(size_t)row * N + colB];
                outf[(size_t)row * N + colB] = v1;
            }
        }
    }
}

// ---------------------------------------------------------------------------
// Causal attention, one (b,h) per 8-wave block. T=64, HD=32. f16 in/out.
// Q/K staged via async b128; V transposed in LDS for contiguous B fragments.
// scores via WMMA (K=HD=32), causal softmax f32, O = att @ V (2 WMMA chain).
// ---------------------------------------------------------------------------
__global__ __launch_bounds__(256) void attention_kernel(const _Float16* __restrict__ Q,
                                                        const _Float16* __restrict__ Km,
                                                        const _Float16* __restrict__ Vm,
                                                        _Float16* __restrict__ O,
                                                        float scale) {
    __shared__ __align__(16) _Float16 sQ[T_ * HD_];     // [t][hd]
    __shared__ __align__(16) _Float16 sK[T_ * HD_];     // [t][hd] (= B^T layout)
    __shared__ __align__(16) _Float16 sVt[HD_ * T_];    // [hd][t] transposed
    __shared__ __align__(16) float    sS[T_ * T_];
    __shared__ __align__(16) _Float16 sAtt[T_ * T_];

    int bh = blockIdx.x;
    int b = bh >> 2;       // H_ = 4
    int h = bh & 3;
    int tid = threadIdx.x;
    int lane = tid & 31;
    int w = tid >> 5;
    size_t base = (size_t)b * T_ * D_ + (size_t)h * HD_;

    // stage: thread owns 8 halves (row tr, cols tc..tc+7)
    {
        int tr = tid >> 2, tc = (tid & 3) * 8;
        size_t g = base + (size_t)tr * D_ + tc;
        async_b128(lds_addr(&sQ[tr * HD_ + tc]), Q + g);
        async_b128(lds_addr(&sK[tr * HD_ + tc]), Km + g);
        v8h vv = *(const v8h*)(Vm + g);
#pragma unroll
        for (int i = 0; i < 8; ++i)
            sVt[(tc + i) * T_ + tr] = vv[i];
        wait_async0();
    }
    __syncthreads();

    // scores: 16 tiles of 16x16, two per wave
#pragma unroll
    for (int rep = 0; rep < 2; ++rep) {
        int tt = w + rep * 8;
        int mi = tt >> 2, ni = tt & 3;
        v16h a  = load_frag_a(sQ, mi * 16, HD_, lane);
        v16h bb = load_frag_bT(sK, ni * 16, HD_, lane);
        v8f acc = {};
        acc = __builtin_amdgcn_wmma_f32_16x16x32_f16(false, a, false, bb,
                                                     (short)0, acc, false, false);
        int cb = ni * 16 + (lane & 15);
        int rb = mi * 16 + ((lane >> 4) << 3);
#pragma unroll
        for (int j = 0; j < 8; ++j) sS[(rb + j) * T_ + cb] = acc[j] * scale;
    }
    __syncthreads();

    // causal softmax: 4 threads per row, 16 cols each
    {
        int r = tid >> 2, sub = tid & 3;
        float vals[16];
        float mx = -3.0e38f;
#pragma unroll
        for (int i = 0; i < 16; ++i) {
            int c = sub * 16 + i;
            float v = (c <= r) ? sS[r * T_ + c] : -1.0e30f;
            vals[i] = v;
            mx = fmaxf(mx, v);
        }
        mx = fmaxf(mx, __shfl_xor(mx, 1, 32));
        mx = fmaxf(mx, __shfl_xor(mx, 2, 32));
        float sum = 0.0f;
#pragma unroll
        for (int i = 0; i < 16; ++i) {
            vals[i] = __expf(vals[i] - mx);
            sum += vals[i];
        }
        sum += __shfl_xor(sum, 1, 32);
        sum += __shfl_xor(sum, 2, 32);
        float inv = 1.0f / sum;
#pragma unroll
        for (int i = 0; i < 16; ++i)
            sAtt[r * T_ + sub * 16 + i] = (_Float16)(vals[i] * inv);
    }
    __syncthreads();

    // O = att(64x64) @ V(64x32): one 16x16 tile per wave, K = 64 = 2 steps
    int mi = w >> 1, ni = w & 1;
    v8f acc = {};
#pragma unroll
    for (int ks = 0; ks < T_; ks += 32) {
        v16h a  = load_frag_a(sAtt + ks, mi * 16, T_, lane);
        v16h bb = load_frag_bT(sVt + ks, ni * 16, T_, lane);
        acc = __builtin_amdgcn_wmma_f32_16x16x32_f16(false, a, false, bb,
                                                     (short)0, acc, false, false);
    }
    int cb = ni * 16 + (lane & 15);
    int rb = mi * 16 + ((lane >> 4) << 3);
#pragma unroll
    for (int j = 0; j < 8; ++j)
        O[base + (size_t)(rb + j) * D_ + cb] = (_Float16)acc[j];
}

// ---------------------------------------------------------------------------
// Loss: per-row log-softmax over V=2000, -mean(logp[target]) via atomicAdd.
// ---------------------------------------------------------------------------
__global__ void zero_kernel(float* p) { *p = 0.0f; }

__global__ __launch_bounds__(256) void loss_kernel(const float* __restrict__ logits,
                                                   const int* __restrict__ targets,
                                                   float* __restrict__ loss) {
    __shared__ float redmax[8];
    __shared__ float redsum[8];
    int row = blockIdx.x;
    int tid = threadIdx.x;
    int lane = tid & 31, w = tid >> 5;
    const float* lr = logits + (size_t)row * V_;

    float mx = -3.0e38f;
    for (int c = tid; c < V_; c += 256) mx = fmaxf(mx, lr[c]);
#pragma unroll
    for (int off = 16; off > 0; off >>= 1) mx = fmaxf(mx, __shfl_xor(mx, off, 32));
    if (lane == 0) redmax[w] = mx;
    __syncthreads();
    float rowmax = redmax[0];
#pragma unroll
    for (int i = 1; i < 8; ++i) rowmax = fmaxf(rowmax, redmax[i]);

    float sum = 0.0f;
    for (int c = tid; c < V_; c += 256) sum += __expf(lr[c] - rowmax);
#pragma unroll
    for (int off = 16; off > 0; off >>= 1) sum += __shfl_xor(sum, off, 32);
    if (lane == 0) redsum[w] = sum;
    __syncthreads();

    if (tid == 0) {
        float s = 0.0f;
#pragma unroll
        for (int i = 0; i < 8; ++i) s += redsum[i];
        int tgt = targets[row];
        float lp = lr[tgt] - rowmax - __logf(s);
        atomicAdd(loss, -lp * (1.0f / (float)BT_));
    }
}

// ---------------------------------------------------------------------------
// Host launcher
// ---------------------------------------------------------------------------
extern "C" void kernel_launch(void* const* d_in, const int* in_sizes, int n_in,
                              void* d_out, int out_size, void* d_ws, size_t ws_size,
                              hipStream_t stream) {
    const int*   idx     = (const int*)d_in[0];
    const int*   targets = (const int*)d_in[1];
    const float* tok_emb = (const float*)d_in[2];
    const float* pos_emb = (const float*)d_in[3];
    const float* Wq      = (const float*)d_in[4];
    const float* Wk      = (const float*)d_in[5];
    const float* Wv      = (const float*)d_in[6];
    const float* Wo      = (const float*)d_in[7];
    const float* bo      = (const float*)d_in[8];
    const float* ln1_g   = (const float*)d_in[9];
    const float* ln1_b   = (const float*)d_in[10];
    const float* W1      = (const float*)d_in[11];
    const float* b1      = (const float*)d_in[12];
    const float* W2      = (const float*)d_in[13];
    const float* b2      = (const float*)d_in[14];
    const float* ln2_g   = (const float*)d_in[15];
    const float* ln2_b   = (const float*)d_in[16];
    const float* lnf_g   = (const float*)d_in[17];
    const float* lnf_b   = (const float*)d_in[18];
    const float* Wh      = (const float*)d_in[19];
    const float* bh      = (const float*)d_in[20];

    const size_t nBT_D = (size_t)BT_ * D_;
    const size_t nBT_F = (size_t)BT_ * FF_;

    float*     x   = (float*)d_ws;                       // f32 residual stream
    _Float16*  h16 = (_Float16*)(x + nBT_D);
    _Float16*  q16 = h16 + nBT_D;
    _Float16*  k16 = q16 + nBT_D;
    _Float16*  v16 = k16 + nBT_D;
    _Float16*  o16 = v16 + nBT_D;
    _Float16*  f16 = o16 + nBT_D;                        // BT x FF
    // pre-transposed f16 weights
    _Float16*  wqt = f16 + nBT_F;                        // L * D*D
    _Float16*  wkt = wqt + (size_t)L_ * D_ * D_;
    _Float16*  wvt = wkt + (size_t)L_ * D_ * D_;
    _Float16*  wot = wvt + (size_t)L_ * D_ * D_;
    _Float16*  w1t = wot + (size_t)L_ * D_ * D_;         // L * FF*D ([FF,D])
    _Float16*  w2t = w1t + (size_t)L_ * D_ * FF_;        // L * D*FF ([D,FF])
    _Float16*  wht = w2t + (size_t)L_ * D_ * FF_;        // V*D ([V,D])

    float* logits = (float*)d_out;
    float* loss   = logits + (size_t)BT_ * V_;

    const float scale = 1.0f / sqrtf((float)D_);   // reference: 1/sqrt(D)

    dim3 blk(256);
    dim3 gLN(BT_ / 8);
    dim3 gD(BT_ / 64, D_ / 64);          // N=128 GEMMs
    dim3 gFF(BT_ / 64, FF_ / 64);        // N=512 GEMM
    dim3 gHead(BT_ / 64, (V_ + 63) / 64);// N=2000 GEMM (edge tiles masked)

    // ---- weight prep (tiny) ----
    for (int l = 0; l < L_; ++l) {
        int gdd = (D_ * D_ + 255) / 256, gdf = (D_ * FF_ + 255) / 256;
        transpose_cvt_kernel<<<gdd, blk, 0, stream>>>(Wq + (size_t)l * D_ * D_,
                                                      wqt + (size_t)l * D_ * D_, D_, D_);
        transpose_cvt_kernel<<<gdd, blk, 0, stream>>>(Wk + (size_t)l * D_ * D_,
                                                      wkt + (size_t)l * D_ * D_, D_, D_);
        transpose_cvt_kernel<<<gdd, blk, 0, stream>>>(Wv + (size_t)l * D_ * D_,
                                                      wvt + (size_t)l * D_ * D_, D_, D_);
        transpose_cvt_kernel<<<gdd, blk, 0, stream>>>(Wo + (size_t)l * D_ * D_,
                                                      wot + (size_t)l * D_ * D_, D_, D_);
        transpose_cvt_kernel<<<gdf, blk, 0, stream>>>(W1 + (size_t)l * D_ * FF_,
                                                      w1t + (size_t)l * D_ * FF_, D_, FF_);
        transpose_cvt_kernel<<<gdf, blk, 0, stream>>>(W2 + (size_t)l * D_ * FF_,
                                                      w2t + (size_t)l * D_ * FF_, FF_, D_);
    }
    transpose_cvt_kernel<<<(D_ * V_ + 255) / 256, blk, 0, stream>>>(Wh, wht, D_, V_);

    // ---- forward ----
    embed_kernel<<<BT_ / 2, blk, 0, stream>>>(idx, tok_emb, pos_emb, x);

    for (int l = 0; l < L_; ++l) {
        const _Float16* wqt_l = wqt + (size_t)l * D_ * D_;
        const _Float16* wkt_l = wkt + (size_t)l * D_ * D_;
        const _Float16* wvt_l = wvt + (size_t)l * D_ * D_;
        const _Float16* wot_l = wot + (size_t)l * D_ * D_;
        const _Float16* w1t_l = w1t + (size_t)l * D_ * FF_;
        const _Float16* w2t_l = w2t + (size_t)l * D_ * FF_;
        const float* bo_l = bo + (size_t)l * D_;
        const float* b1_l = b1 + (size_t)l * FF_;
        const float* b2_l = b2 + (size_t)l * D_;

        layernorm_kernel<<<gLN, blk, 0, stream>>>(x, ln1_g + (size_t)l * D_,
                                                  ln1_b + (size_t)l * D_, h16);

        gemm_wmma_kernel<<<gD, blk, 0, stream>>>(h16, wqt_l, nullptr, nullptr,
                                                 nullptr, q16, BT_, D_, D_, 0);
        gemm_wmma_kernel<<<gD, blk, 0, stream>>>(h16, wkt_l, nullptr, nullptr,
                                                 nullptr, k16, BT_, D_, D_, 0);
        gemm_wmma_kernel<<<gD, blk, 0, stream>>>(h16, wvt_l, nullptr, nullptr,
                                                 nullptr, v16, BT_, D_, D_, 0);

        attention_kernel<<<dim3(1024 * H_), blk, 0, stream>>>(q16, k16, v16, o16,
                                                              scale);

        // x = x + o @ Wo + bo   (f32 out + residual)
        gemm_wmma_kernel<<<gD, blk, 0, stream>>>(o16, wot_l, bo_l, x,
                                                 x, nullptr, BT_, D_, D_, 0);

        layernorm_kernel<<<gLN, blk, 0, stream>>>(x, ln2_g + (size_t)l * D_,
                                                  ln2_b + (size_t)l * D_, h16);

        // ff = relu(h @ W1 + b1)  (f16 out)
        gemm_wmma_kernel<<<gFF, blk, 0, stream>>>(h16, w1t_l, b1_l, nullptr,
                                                  nullptr, f16, BT_, FF_, D_, 1);
        // x = x + ff @ W2 + b2   (f32 out + residual)
        gemm_wmma_kernel<<<gD, blk, 0, stream>>>(f16, w2t_l, b2_l, x,
                                                 x, nullptr, BT_, D_, FF_, 0);
    }

    layernorm_kernel<<<gLN, blk, 0, stream>>>(x, lnf_g, lnf_b, h16);

    // logits = h @ Wh + bh  (streams 524 MB to HBM -> the real bottleneck)
    gemm_wmma_kernel<<<gHead, blk, 0, stream>>>(h16, wht, bh, nullptr,
                                                logits, nullptr, BT_, V_, D_, 0);

    zero_kernel<<<1, 1, 0, stream>>>(loss);
    loss_kernel<<<dim3(BT_), blk, 0, stream>>>(logits, targets, loss);
}
